// OffGridPatchEmbed_19387482374262
// MI455X (gfx1250) — compile-verified, matrix operand order checked by reading, not run on hardware
//
#include <hip/hip_runtime.h>
#include <hip/hip_bf16.h>
#include <stdint.h>

// ---------------------------------------------------------------------------
// OffGridPatchEmbed for MI455X (gfx1250, wave32, WMMA).
//
// patch_tokens[b,n,d] = sum_f patch[b,n,f] * W_proj[d,f] + b_proj[d]
//   M = B*N = 25088 rows, K = F = 768, Ncols = D = 768
//
// f32 GEMM executed as bf16x3 split on v_wmma_f32_16x16x32_bf16:
//   A = Ahi + Alo;  D += Ahi*Bhi + Alo*Bhi + Ahi*Blo  (f32 acc, ~2^-16 rel)
//
// v2: 32 rows (2 M-tiles) per workgroup -> each B fragment feeds 6 WMMAs,
//     halving L2 W-traffic and doubling FLOP/byte vs v1. B addresses are
//     uniform-base + 32-bit offset with constant tile/k immediates.
// ---------------------------------------------------------------------------

typedef __attribute__((ext_vector_type(16))) __bf16 v16bf;
typedef __attribute__((ext_vector_type(8)))  float  v8f;
typedef unsigned short u16;

#define Bn    128
#define Cn    3
#define Hn    224
#define Wn    224
#define Nn    196
#define Fn    768            // C*P*P (= K)
#define Dn    768
#define MTOT  (Bn * Nn)      // 25088 (multiple of 32)
#define KSTEPS (Fn / 32)     // 24
#define DTILES (Dn / 16)     // 48
#define FRAG_PER_KS   (32 * 16)                    // 512 elems per (dt,ks)
#define FRAG_PER_DT   (KSTEPS * FRAG_PER_KS)       // 12288 elems per dt
#define WFRAG_ELEMS   (DTILES * FRAG_PER_DT)       // 589824 bf16 per array
#define ROWTILE_ELEMS (16 * Fn)                    // 12288 u16 per row-tile

// f32 -> (hi,lo) bf16 split. hi = truncate, lo = RNE(v - hi).
__device__ __forceinline__ void bf16_split(float v, u16& hi, u16& lo) {
    uint32_t u = __float_as_uint(v);
    hi = (u16)(u >> 16);
    float lof = v - __uint_as_float(((uint32_t)hi) << 16);
    uint32_t ul = __float_as_uint(lof);
    lo = (u16)((ul + 0x7FFFu + ((ul >> 16) & 1u)) >> 16);
}

// -------- Kernel 0: pre-split W_proj into WMMA B-fragment layout ------------
// 16-bit fragment layout (ISA 7.12.2): lane l holds column n = l&15,
// element j (0..15) holds K = (j<8 ? j : j+8) + (l<16 ? 0 : 8) within a
// 32-wide K step. Each lane's 16 bf16 are 32B-contiguous.
__global__ __launch_bounds__(256)
void prep_w_kernel(const float* __restrict__ Wp,
                   u16* __restrict__ Whi, u16* __restrict__ Wlo) {
    int idx = blockIdx.x * 256 + threadIdx.x;       // < WFRAG_ELEMS
    int j  = idx & 15;
    int l  = (idx >> 4) & 31;
    int t  = idx >> 9;                               // dt*KSTEPS + ks
    int ks = t % KSTEPS;
    int dt = t / KSTEPS;
    int d  = dt * 16 + (l & 15);
    int k  = ks * 32 + ((j < 8) ? j : j + 8) + ((l < 16) ? 0 : 8);
    u16 hi, lo;
    bf16_split(Wp[d * Fn + k], hi, lo);
    Whi[idx] = hi;
    Wlo[idx] = lo;
}

// -------- Kernel 1: positions output (float, exact small ints) --------------
__global__ __launch_bounds__(256)
void pos_kernel(const int* __restrict__ ys, const int* __restrict__ xs,
                float* __restrict__ pos) {
    int i = blockIdx.x * 256 + threadIdx.x;
    if (i < MTOT) {
        pos[2 * i + 0] = (float)ys[i];
        pos[2 * i + 1] = (float)xs[i];
    }
}

// -------- Kernel 2: fused gather + GEMM -------------------------------------
// WG = 32 rows x 768 cols, 256 threads = 8 waves.
// Wave w owns col tiles d0 = w*16 + i*128 (i = 0..5) for BOTH row tiles.
extern __shared__ __align__(32) char smem[];

__global__ __launch_bounds__(256)
void patch_embed_kernel(const float* __restrict__ x,
                        const int* __restrict__ ys, const int* __restrict__ xs,
                        const u16* __restrict__ Whi, const u16* __restrict__ Wlo,
                        const float* __restrict__ bias,
                        float* __restrict__ out) {
    u16* sHi  = (u16*)smem;                               // 2 row-tiles, 48 KB
    u16* sLo  = sHi + 2 * ROWTILE_ELEMS;                  // 48 KB
    int* sOff = (int*)(sLo + 2 * ROWTILE_ELEMS);          // 32 ints

    const int tid = threadIdx.x;
    const int rowbase = blockIdx.x * 32;

    // Per-row base offsets into x (channel 0, dy=dx=0).
    if (tid < 32) {
        int row = rowbase + tid;
        int b = row / Nn;                          // ys/xs flat index == row
        sOff[tid] = b * (Cn * Hn * Wn) + ys[row] * Wn + xs[row];
    }
    __syncthreads();

    // Gather 32 rows x 768 features; split to bf16 hi/lo; scatter into the
    // A-fragment layout [rt][ks][lane][j]. 768 = 3*256 -> coalesced 16-wide
    // dx runs per 256-thread sweep.
    for (int m = 0; m < 32; ++m) {
        const int base = sOff[m];
        const int rt   = m >> 4;
        const int mm   = m & 15;
#pragma unroll
        for (int t3 = 0; t3 < 3; ++t3) {
            int k = t3 * 256 + tid;
            int c = k >> 8;
            int p = k & 255;
            float v = x[base + c * (Hn * Wn) + (p >> 4) * Wn + (p & 15)];
            u16 hi, lo;
            bf16_split(v, hi, lo);
            // inverse of the fragment K mapping
            int kk = k & 31;
            int ks = k >> 5;
            int lane = mm + ((((kk >> 3) & 1) != 0) ? 16 : 0);
            int j = (kk & 7) + ((kk >= 16) ? 8 : 0);
            int sidx = rt * ROWTILE_ELEMS + (ks * 32 + lane) * 16 + j;
            sHi[sidx] = hi;
            sLo[sidx] = lo;
        }
    }
    __syncthreads();

    const int wave = tid >> 5;
    const int lane = tid & 31;

    // Uniform-base + 32-bit offset B addressing: (i, ks) become constant
    // immediates folded into the load offset field.
    const unsigned boff = (unsigned)(wave * FRAG_PER_DT + lane * 16);
    const u16* __restrict__ pWhi = Whi + boff;
    const u16* __restrict__ pWlo = Wlo + boff;
    const unsigned aoff = (unsigned)(lane * 16);

    v8f acc[2][6] = {};

    for (int ks = 0; ks < KSTEPS; ++ks) {
        const unsigned ak = aoff + (unsigned)ks * FRAG_PER_KS;
        const v16bf a0h = *(const v16bf*)&sHi[ak];
        const v16bf a0l = *(const v16bf*)&sLo[ak];
        const v16bf a1h = *(const v16bf*)&sHi[ak + ROWTILE_ELEMS];
        const v16bf a1l = *(const v16bf*)&sLo[ak + ROWTILE_ELEMS];
        const unsigned bk = (unsigned)ks * FRAG_PER_KS;
#pragma unroll
        for (int i = 0; i < 6; ++i) {
            const v16bf bh = *(const v16bf*)(pWhi + bk + i * 8 * FRAG_PER_DT);
            const v16bf bl = *(const v16bf*)(pWlo + bk + i * 8 * FRAG_PER_DT);
            acc[0][i] = __builtin_amdgcn_wmma_f32_16x16x32_bf16(
                false, a0h, false, bh, (short)0, acc[0][i], false, false);
            acc[1][i] = __builtin_amdgcn_wmma_f32_16x16x32_bf16(
                false, a1h, false, bh, (short)0, acc[1][i], false, false);
            acc[0][i] = __builtin_amdgcn_wmma_f32_16x16x32_bf16(
                false, a0l, false, bh, (short)0, acc[0][i], false, false);
            acc[1][i] = __builtin_amdgcn_wmma_f32_16x16x32_bf16(
                false, a1l, false, bh, (short)0, acc[1][i], false, false);
            acc[0][i] = __builtin_amdgcn_wmma_f32_16x16x32_bf16(
                false, a0h, false, bl, (short)0, acc[0][i], false, false);
            acc[1][i] = __builtin_amdgcn_wmma_f32_16x16x32_bf16(
                false, a1h, false, bl, (short)0, acc[1][i], false, false);
        }
    }

    // Epilogue: C/D layout -> VGPR v holds M = v + 8*(lane>=16), N = lane&15.
    const int col  = lane & 15;
    const int mofs = (lane >> 4) * 8;
#pragma unroll
    for (int i = 0; i < 6; ++i) {
        int d = wave * 16 + i * 128 + col;
        float bv = bias[d];
#pragma unroll
        for (int rt = 0; rt < 2; ++rt) {
#pragma unroll
            for (int v = 0; v < 8; ++v) {
                int row = rowbase + rt * 16 + mofs + v;
                out[row * Dn + d] = acc[rt][i][v] + bv;
            }
        }
    }
}

// ---------------------------------------------------------------------------
extern "C" void kernel_launch(void* const* d_in, const int* in_sizes, int n_in,
                              void* d_out, int out_size, void* d_ws, size_t ws_size,
                              hipStream_t stream) {
    const float* x   = (const float*)d_in[0];
    const int*   ys  = (const int*)d_in[1];
    const int*   xs  = (const int*)d_in[2];
    const float* Wp  = (const float*)d_in[3];
    const float* bp  = (const float*)d_in[4];
    float* out = (float*)d_out;

    u16* Whi = (u16*)d_ws;
    u16* Wlo = Whi + WFRAG_ELEMS;   // total ~2.4 MB of workspace

    prep_w_kernel<<<WFRAG_ELEMS / 256, 256, 0, stream>>>(Wp, Whi, Wlo);
    pos_kernel<<<(MTOT + 255) / 256, 256, 0, stream>>>(
        ys, xs, out + (size_t)MTOT * Dn);

    const size_t lds_bytes = 4u * ROWTILE_ELEMS * sizeof(u16)  // sHi + sLo
                           + 32 * sizeof(int);                  // sOff
    patch_embed_kernel<<<MTOT / 32, 256, lds_bytes, stream>>>(
        x, ys, xs, Whi, Wlo, bp, out);
}
